// TextEncoderObj_57234734186513
// MI455X (gfx1250) — matrix operand order repaired and checked by reading, not compiled, not run on hardware
//
#include <hip/hip_runtime.h>

typedef __attribute__((ext_vector_type(16))) _Float16 v16h;
typedef __attribute__((ext_vector_type(8)))  float    v8f;

namespace {
constexpr int kB    = 512;
constexpr int kT5   = 5;
constexpr int kE    = 1024;
constexpr int kWD   = 512;
constexpr int kMF   = 2048;
constexpr int kMH   = 512;
constexpr int kTMAX = 20;
constexpr int kH    = 256;
constexpr int kNOBJ = 36;
constexpr int kGIN  = kWD + kMF + kMH;  // 3072
constexpr int kG3   = 3 * kH;           // 768
constexpr int kKV   = kH + kE;          // 1280
constexpr int kE2   = 2 * kE;           // 2048
}

// ---------------------------------------------------------------------------
// WMMA GEMM: C[M,N] = A[M,K](f16, row-major) @ Bt[N,K](f16)^T + bias
// One wave -> 32(M) x 64(N) tile: 2 A-frags x 4 B-frags -> 8 wmma per K-step.
// Double-buffered fragment prefetch (K must be a multiple of 64).
// Block = 8 waves -> 256(M) x 64(N). Requires M%256==0, N%64==0.
// ---------------------------------------------------------------------------
union Frag16 { v16h v; uint4 u[2]; };

struct FragSet {
  uint4 a[2][2];   // [m-subtile][16B chunk]
  uint4 b[4][2];   // [n-tile][16B chunk]
};

__device__ __forceinline__ void load_frags(FragSet& f,
    const _Float16* __restrict__ aRow0, const _Float16* __restrict__ aRow1,
    const _Float16* const* __restrict__ bRow, int k0)
{
  f.a[0][0] = *(const uint4*)(aRow0 + k0);
  f.a[0][1] = *(const uint4*)(aRow0 + k0 + 16);
  f.a[1][0] = *(const uint4*)(aRow1 + k0);
  f.a[1][1] = *(const uint4*)(aRow1 + k0 + 16);
#pragma unroll
  for (int j = 0; j < 4; ++j) {
    f.b[j][0] = *(const uint4*)(bRow[j] + k0);
    f.b[j][1] = *(const uint4*)(bRow[j] + k0 + 16);
  }
}

__device__ __forceinline__ void mma_frags(v8f acc[2][4], const FragSet& f)
{
#pragma unroll
  for (int s = 0; s < 2; ++s) {
    Frag16 av; av.u[0] = f.a[s][0]; av.u[1] = f.a[s][1];
#pragma unroll
    for (int j = 0; j < 4; ++j) {
      Frag16 bv; bv.u[0] = f.b[j][0]; bv.u[1] = f.b[j][1];
      acc[s][j] = __builtin_amdgcn_wmma_f32_16x16x32_f16(
          false, av.v, false, bv.v, (short)0, acc[s][j], false, false);
    }
  }
}

__global__ __launch_bounds__(256) void wmma_gemm_f16_kernel(
    const _Float16* __restrict__ A,
    const _Float16* __restrict__ Bt,
    float* __restrict__ C,
    const float* __restrict__ bias,
    int M, int N, int K)
{
  const int lane  = threadIdx.x;       // 0..31
  const int wave  = threadIdx.y;       // 0..7
  const int l15   = lane & 15;
  const int hi    = lane >> 4;         // 0 or 1
  const int nBase = blockIdx.x * 64;
  const int mWave = (blockIdx.y * 8 + wave) * 32;
  if (mWave >= M || nBase >= N) return;

  // A fragment rows (two 16-row subtiles per wave)
  const _Float16* aRow0 = A + (size_t)(mWave + l15) * K + hi * 8;
  const _Float16* aRow1 = aRow0 + (size_t)16 * K;
  // B fragment rows: weight row n = nBase + j*16 + l15, two 8-half chunks (k = 8h.., 16+8h..)
  const _Float16* bRow[4];
#pragma unroll
  for (int j = 0; j < 4; ++j)
    bRow[j] = Bt + (size_t)(nBase + j * 16 + l15) * K + hi * 8;

  v8f acc[2][4];
#pragma unroll
  for (int s = 0; s < 2; ++s)
#pragma unroll
    for (int j = 0; j < 4; ++j)
      acc[s][j] = v8f{};

  FragSet f0, f1;
  load_frags(f0, aRow0, aRow1, bRow, 0);

  for (int k0 = 0; k0 < K; k0 += 64) {
    load_frags(f1, aRow0, aRow1, bRow, k0 + 32);   // prefetch (in-flight)
    mma_frags(acc, f0);                            // compute current
    const int kn = (k0 + 64 < K) ? (k0 + 64) : 0;  // safe dummy on last iter
    load_frags(f0, aRow0, aRow1, bRow, kn);        // prefetch (in-flight)
    mma_frags(acc, f1);                            // compute current
  }

#pragma unroll
  for (int s = 0; s < 2; ++s) {
#pragma unroll
    for (int j = 0; j < 4; ++j) {
      const int n   = nBase + j * 16 + l15;
      const float bv = bias ? bias[n] : 0.0f;
      float* cBase = C + (size_t)(mWave + s * 16 + hi * 8) * N + n;
#pragma unroll
      for (int v = 0; v < 8; ++v)
        cBase[(size_t)v * N] = acc[s][j][v] + bv;
    }
  }
}

// ---------------------------------------------------------------------------
// Stage A: masked object mean. One block per batch element.
// ---------------------------------------------------------------------------
__global__ __launch_bounds__(256) void meanimfeats_kernel(
    const float* __restrict__ obj, float* __restrict__ out)
{
  __shared__ float red[256];
  __shared__ float m[kNOBJ];
  __shared__ float denom;
  const int b = blockIdx.x, tid = threadIdx.x;
  const float* base = obj + (size_t)b * kNOBJ * kMF;

  for (int o = 0; o < kNOBJ; ++o) {
    float s = 0.f;
    for (int f = tid; f < kMF; f += 256) s += fabsf(base[(size_t)o * kMF + f]);
    red[tid] = s; __syncthreads();
    for (int st = 128; st > 0; st >>= 1) {
      if (tid < st) red[tid] += red[tid + st];
      __syncthreads();
    }
    if (tid == 0) m[o] = (red[0] > 0.f) ? 1.f : 0.f;
    __syncthreads();
  }
  if (tid == 0) {
    float d = 0.f;
    for (int o = 0; o < kNOBJ; ++o) d += m[o];
    denom = fmaxf(d, 1e-8f);
  }
  __syncthreads();
  for (int f = tid; f < kMF; f += 256) {
    float acc = 0.f;
    for (int o = 0; o < kNOBJ; ++o) acc += base[(size_t)o * kMF + f] * m[o];
    out[(size_t)b * kMF + f] = acc / denom;
  }
}

// f32 -> f16 with optional row slicing (dst[r,c] = src[r, colofs+c])
__global__ void cvt_slice_f16_kernel(const float* __restrict__ src,
                                     _Float16* __restrict__ dst,
                                     int rows, int srcld, int colofs, int cols)
{
  long long i = (long long)blockIdx.x * blockDim.x + threadIdx.x;
  long long n = (long long)rows * cols;
  if (i >= n) return;
  int r = (int)(i / cols), c = (int)(i % cols);
  dst[i] = (_Float16)src[(size_t)r * srcld + colofs + c];
}

__global__ void zero_f16_kernel(_Float16* __restrict__ p, long long n)
{
  long long i = (long long)blockIdx.x * blockDim.x + threadIdx.x;
  if (i < n) p[i] = (_Float16)0.0f;
}

// GRU input: X[b*T5+t] = [emb[x_obj[b,t]], meanimfeats[b], mask_hidden[b]]
__global__ __launch_bounds__(256) void build_x_kernel(
    const int* __restrict__ x_obj, const float* __restrict__ emb,
    const float* __restrict__ meanimf, const float* __restrict__ mask_hidden,
    _Float16* __restrict__ X)
{
  const int r = blockIdx.x;           // 0..2559
  const int b = r / kT5;
  const int idx = x_obj[r];
  _Float16* xr = X + (size_t)r * kGIN;
  for (int j = threadIdx.x; j < kGIN; j += 256) {
    float v;
    if (j < kWD)            v = emb[(size_t)idx * kWD + j];
    else if (j < kWD + kMF) v = meanimf[(size_t)b * kMF + (j - kWD)];
    else                    v = mask_hidden[(size_t)b * kMH + (j - kWD - kMF)];
    xr[j] = (_Float16)v;
  }
}

// GRU gate fusion for step t. hprev==nullptr means h0 = 0.
__global__ __launch_bounds__(256) void gru_gate_kernel(
    const float* __restrict__ GI, const float* __restrict__ GH,
    const float* __restrict__ hprev, float* __restrict__ hout,
    _Float16* __restrict__ hf16, int t)
{
  const int b = blockIdx.x, j = threadIdx.x;   // j < 256 == kH
  const float* gi = GI + ((size_t)b * kT5 + t) * kG3;
  const float* gh = GH + (size_t)b * kG3;
  const float hp = hprev ? hprev[(size_t)b * kH + j] : 0.f;
  const float r  = 1.f / (1.f + expf(-(gi[j]        + gh[j])));
  const float z  = 1.f / (1.f + expf(-(gi[kH + j]   + gh[kH + j])));
  const float n  = tanhf(gi[2 * kH + j] + r * gh[2 * kH + j]);
  const float hn = (1.f - z) * n + z * hp;
  hout[(size_t)b * kH + j] = hn;
  hf16[(size_t)b * kH + j] = (_Float16)hn;
}

// hidden_o[b] = sum_{t<len} h_t[b] / len
__global__ __launch_bounds__(256) void hidden_kernel(
    const float* __restrict__ Hall, const int* __restrict__ lengths,
    float* __restrict__ hidden)
{
  const int b = blockIdx.x, j = threadIdx.x;
  const int len = lengths[b];
  float s = 0.f;
  for (int t = 0; t < len; ++t)
    s += Hall[((size_t)t * kB + b) * kH + j];
  hidden[(size_t)b * kH + j] = s / (float)len;
}

// Static VAE input: Xv[b*TMAX+t] = [hidden_o[b], mask_z[b,t]]
__global__ __launch_bounds__(256) void build_xv_kernel(
    const float* __restrict__ hidden, const float* __restrict__ mask_z,
    _Float16* __restrict__ Xv)
{
  const int r = blockIdx.x;            // 0..10239
  const int b = r / kTMAX;
  _Float16* xr = Xv + (size_t)r * kKV;
  for (int j = threadIdx.x; j < kKV; j += 256) {
    float v = (j < kH) ? hidden[(size_t)b * kH + j]
                       : mask_z[(size_t)r * kE + (j - kH)];
    xr[j] = (_Float16)v;
  }
}

// VAE step t: o = Ostat[b,t] + Odyn[b]; mean/log split; z = mean + eps*exp(.5*log)
__global__ __launch_bounds__(256) void vae_gate_kernel(
    const float* __restrict__ Ostat, const float* __restrict__ Odyn,
    const float* __restrict__ noise, float* __restrict__ outm,
    float* __restrict__ outl, float* __restrict__ outz,
    _Float16* __restrict__ zf16, int t)
{
  const int b = blockIdx.x;
  const float* os = Ostat + ((size_t)b * kTMAX + t) * kE2;
  const float* od = Odyn + (size_t)b * kE2;
  const size_t ro = ((size_t)b * kTMAX + t) * kE;
  for (int j = threadIdx.x; j < kE; j += 256) {
    const float mean = os[j] + od[j];
    const float lg   = os[kE + j] + od[kE + j];
    const float zv   = mean + noise[ro + j] * expf(0.5f * lg);
    outm[ro + j] = mean;
    outl[ro + j] = lg;
    outz[ro + j] = zv;
    zf16[(size_t)b * kE + j] = (_Float16)zv;
  }
}

// ---------------------------------------------------------------------------
static inline void launch_gemm(const _Float16* A, const _Float16* Bt, float* C,
                               const float* bias, int M, int N, int K,
                               hipStream_t s)
{
  dim3 grid(N / 64, M / 256), block(32, 8);
  hipLaunchKernelGGL(wmma_gemm_f16_kernel, grid, block, 0, s, A, Bt, C, bias, M, N, K);
}

extern "C" void kernel_launch(void* const* d_in, const int* in_sizes, int n_in,
                              void* d_out, int out_size, void* d_ws, size_t ws_size,
                              hipStream_t stream)
{
  (void)in_sizes; (void)n_in; (void)out_size; (void)ws_size;

  const int*   x_obj       = (const int*)d_in[0];
  const float* obj_enc     = (const float*)d_in[1];
  const float* mask_hidden = (const float*)d_in[2];
  const float* mask_z      = (const float*)d_in[3];
  const int*   lengths     = (const int*)d_in[4];
  const float* noise       = (const float*)d_in[5];
  const float* emb         = (const float*)d_in[6];
  const float* w_ih        = (const float*)d_in[7];
  const float* w_hh        = (const float*)d_in[8];
  const float* b_ih        = (const float*)d_in[9];
  const float* b_hh        = (const float*)d_in[10];
  const float* fc_w        = (const float*)d_in[11];
  const float* fc_b        = (const float*)d_in[12];

  float* out_means = (float*)d_out;
  const size_t OUT1 = (size_t)kB * kTMAX * kE;   // 10,485,760
  float* out_logs  = out_means + OUT1;
  float* out_z     = out_means + 2 * OUT1;

  // ---- workspace partition (256B aligned) ----
  char* p = (char*)d_ws;
  auto take = [&](size_t bytes) -> char* {
    char* r = p;
    p += (bytes + 255) & ~(size_t)255;
    return r;
  };
  float*    meanimf  = (float*)   take((size_t)kB * kMF * 4);
  _Float16* w_ih_h   = (_Float16*)take((size_t)kG3 * kGIN * 2);
  _Float16* w_hh_h   = (_Float16*)take((size_t)kG3 * kH * 2);
  _Float16* fc_ws_h  = (_Float16*)take((size_t)kE2 * kKV * 2);   // fc_w[:, :1280]
  _Float16* fc_wz_h  = (_Float16*)take((size_t)kE2 * kE * 2);    // fc_w[:, 1280:]
  _Float16* Xf16     = (_Float16*)take((size_t)kB * kT5 * kGIN * 2);
  float*    GI       = (float*)   take((size_t)kB * kT5 * kG3 * 4);
  float*    GH       = (float*)   take((size_t)kB * kG3 * 4);
  float*    Hall     = (float*)   take((size_t)kT5 * kB * kH * 4);
  _Float16* h_f16    = (_Float16*)take((size_t)kB * kH * 2);
  float*    hidden   = (float*)   take((size_t)kB * kH * 4);
  _Float16* Xv       = (_Float16*)take((size_t)kB * kTMAX * kKV * 2);
  float*    Ostat    = (float*)   take((size_t)kB * kTMAX * kE2 * 4);
  float*    Odyn     = (float*)   take((size_t)kB * kE2 * 4);
  _Float16* z_f16    = (_Float16*)take((size_t)kB * kE * 2);

  // ---- Stage A: masked object mean (memory-bound) ----
  hipLaunchKernelGGL(meanimfeats_kernel, dim3(kB), dim3(256), 0, stream,
                     obj_enc, meanimf);

  // ---- weight conversions to f16 ----
  auto cvt = [&](const float* src, _Float16* dst, int rows, int srcld,
                 int colofs, int cols) {
    long long n = (long long)rows * cols;
    hipLaunchKernelGGL(cvt_slice_f16_kernel,
                       dim3((unsigned)((n + 255) / 256)), dim3(256), 0, stream,
                       src, dst, rows, srcld, colofs, cols);
  };
  cvt(w_ih, w_ih_h, kG3, kGIN, 0, kGIN);
  cvt(w_hh, w_hh_h, kG3, kH, 0, kH);
  cvt(fc_w, fc_ws_h, kE2, kH + 2 * kE, 0, kKV);      // [:, 0:1280]
  cvt(fc_w, fc_wz_h, kE2, kH + 2 * kE, kKV, kE);     // [:, 1280:2304]

  // ---- GRU input build + hoisted input GEMM (all 5 steps at once) ----
  hipLaunchKernelGGL(build_x_kernel, dim3(kB * kT5), dim3(256), 0, stream,
                     x_obj, emb, meanimf, mask_hidden, Xf16);
  launch_gemm(Xf16, w_ih_h, GI, b_ih, kB * kT5, kG3, kGIN, stream);  // 2560x3072x768

  // ---- sequential GRU recurrence (tiny GEMM + gate fusion per step) ----
  {
    long long n = (long long)kB * kH;
    hipLaunchKernelGGL(zero_f16_kernel, dim3((unsigned)((n + 255) / 256)),
                       dim3(256), 0, stream, h_f16, n);
  }
  for (int t = 0; t < kT5; ++t) {
    launch_gemm(h_f16, w_hh_h, GH, b_hh, kB, kG3, kH, stream);       // 512x256x768
    const float* hprev = (t == 0) ? nullptr : (Hall + (size_t)(t - 1) * kB * kH);
    hipLaunchKernelGGL(gru_gate_kernel, dim3(kB), dim3(kH), 0, stream,
                       GI, GH, hprev, Hall + (size_t)t * kB * kH, h_f16, t);
  }
  hipLaunchKernelGGL(hidden_kernel, dim3(kB), dim3(kH), 0, stream,
                     Hall, lengths, hidden);

  // ---- VAE: hoisted static GEMM for all 20 steps ----
  hipLaunchKernelGGL(build_xv_kernel, dim3(kB * kTMAX), dim3(256), 0, stream,
                     hidden, mask_z, Xv);
  launch_gemm(Xv, fc_ws_h, Ostat, fc_b, kB * kTMAX, kE2, kKV, stream); // 10240x1280x2048

  // ---- sequential VAE recurrence ----
  {
    long long n = (long long)kB * kE;
    hipLaunchKernelGGL(zero_f16_kernel, dim3((unsigned)((n + 255) / 256)),
                       dim3(256), 0, stream, z_f16, n);
  }
  for (int t = 0; t < kTMAX; ++t) {
    launch_gemm(z_f16, fc_wz_h, Odyn, nullptr, kB, kE2, kE, stream);  // 512x1024x2048
    hipLaunchKernelGGL(vae_gate_kernel, dim3(kB), dim3(256), 0, stream,
                       Ostat, Odyn, noise, out_means, out_logs, out_z, z_f16, t);
  }
}
